// Multibin_81990925681322
// MI455X (gfx1250) — compile-verified, matrix operand order checked by reading, not run on hardware
//
#include <hip/hip_runtime.h>
#include <stdint.h>

typedef __attribute__((ext_vector_type(16))) _Float16 v16h;
typedef __attribute__((ext_vector_type(8)))  _Float16 v8h;
typedef __attribute__((ext_vector_type(8)))  float    v8f;

#define IMG_H 256
#define IMG_W 256
#define CIN   32
#define COUT  32
#define NTAPS 9

// Halo tile: (16+2) x (16+2) pixels, 32 channels of f16 signs per pixel.
// Pixel stride padded to 40 halves (80 B): 16B-aligned for ds_load_b128 and a
// 20-bank step across lanes (20*d % 64 != 0 for d=1..15 -> conflict-free).
#define HALO         18
#define PIX_STRIDE_H 40                            // halves per pixel slot
#define LDS_HALVES   (HALO * HALO * PIX_STRIDE_H)  // 12960 halves = 25920 B

// ---------------------------------------------------------------------------
// Prep kernel: w_sum = sum_m sign(W[m]) in WMMA B-fragment order + bias sums.
// wsB layout (halves): idx = ((tap*2 + ntile)*32 + lane)*16 + slot
//   K(lane,slot)  = (lane>=16 ? 8 : 0) + (slot<8 ? slot : slot+8)
//   Cout(lane,nt) = nt*16 + (lane & 15)
// ---------------------------------------------------------------------------
__global__ void multibin_prep(const float* __restrict__ W,
                              const float* __restrict__ bias,
                              _Float16* __restrict__ wsB,
                              float* __restrict__ bsum) {
    int t = blockIdx.x * blockDim.x + threadIdx.x;
    if (t < NTAPS * 2 * 32 * 16) {
        int slot = t & 15;
        int lane = (t >> 4) & 31;
        int nt   = (t >> 9) & 1;
        int tap  = t >> 10;
        int ky = tap / 3, kx = tap % 3;
        int k  = ((lane >= 16) ? 8 : 0) + (slot < 8 ? slot : slot + 8);
        int co = nt * 16 + (lane & 15);
        float s = 0.f;
#pragma unroll
        for (int m = 0; m < 4; ++m) {
            float w = W[(((m * 3 + ky) * 3 + kx) * CIN + k) * COUT + co];
            s += (w >= 0.f) ? 1.f : -1.f;
        }
        wsB[t] = (_Float16)s;
    }
    if (t < COUT) {
        float s = 0.f;
#pragma unroll
        for (int m = 0; m < 4; ++m) s += bias[m * COUT + t];
        bsum[t] = s;
    }
}

// ---------------------------------------------------------------------------
// Main kernel: 256 threads (8 waves) compute a 16x16 pixel x 32 Cout tile.
// Stage 1: binarize the 18x18x32 halo once into LDS (bit-trick sign).
// Stage 2: 9 taps x 2 row-tiles x 2 N-tiles of v_wmma_f32_16x16x32_f16,
//          with all A addresses folded into ds_load immediate offsets.
// ---------------------------------------------------------------------------
__global__ __launch_bounds__(256)
void multibin_conv(const float* __restrict__ x,
                   const _Float16* __restrict__ wsB,
                   const float* __restrict__ bsum,
                   float* __restrict__ out) {
    __shared__ __align__(16) uint16_t sSign[LDS_HALVES];

    const int img = blockIdx.x >> 8;
    const int h0  = ((blockIdx.x >> 4) & 15) * 16;
    const int w0  = (blockIdx.x & 15) * 16;

    // ---------------- Stage 1: halo binarize into LDS ----------------
    // 18*18 pixels * 8 float4-chunks = 2592 chunks of 4 channels.
    const uint32_t* xb = (const uint32_t*)x;
    for (int i = threadIdx.x; i < HALO * HALO * 8; i += 256) {
        const int pixel = i >> 3;
        const int part  = i & 7;
        const int hh = pixel / HALO;            // 0..17
        const int ww = pixel - hh * HALO;       // 0..17
        const int h  = h0 + hh - 1;
        const int w  = w0 + ww - 1;
        uint32_t p0 = 0u, p1 = 0u;
        if ((unsigned)h < (unsigned)IMG_H && (unsigned)w < (unsigned)IMG_W) {
            const uint32_t* src =
                xb + (((size_t)img * IMG_H + h) * IMG_W + w) * CIN + part * 4;
            const uint32_t u0 = src[0], u1 = src[1], u2 = src[2], u3 = src[3];
            // sign(x>=0 -> +1h, else -1h): 0x3C00 | sign-bit (packed pairs)
            p0 = 0x3C003C00u | ((u0 >> 16) & 0x8000u) | (u1 & 0x80000000u);
            p1 = 0x3C003C00u | ((u2 >> 16) & 0x8000u) | (u3 & 0x80000000u);
        }
        uint32_t* dst = (uint32_t*)(sSign + pixel * PIX_STRIDE_H + part * 4);
        dst[0] = p0;
        dst[1] = p1;
    }
    __syncthreads();

    // ---------------- Stage 2: WMMA over 9 taps ----------------
    const int lane   = threadIdx.x & 31;
    const int wv     = threadIdx.x >> 5;       // 0..7: row-tiles 2wv, 2wv+1
    const int m      = lane & 15;              // A row / D column
    const int laneHi = lane >> 4;              // 0 or 1

    // Single VALU base; every tap becomes a constant ds_load immediate offset:
    //   half-offset(ky,kx,rr) = ((rr+ky)*18 + kx) * PIX_STRIDE_H  (+16 for hi run)
    const uint16_t* aBase = sSign + ((wv * 2) * HALO + m) * PIX_STRIDE_H + laneHi * 8;
    // B fragments: one SGPR base + lane offset; tap/ntile are constant offsets.
    const _Float16* bBase = wsB + (size_t)lane * 16;

    v8f acc00 = {}, acc01 = {}, acc10 = {}, acc11 = {};

#pragma unroll
    for (int ky = 0; ky < 3; ++ky) {
#pragma unroll
        for (int kx = 0; kx < 3; ++kx) {
            const int tap = ky * 3 + kx;
            const v16h b0 = *(const v16h*)(bBase + (size_t)(tap * 2 + 0) * 512);
            const v16h b1 = *(const v16h*)(bBase + (size_t)(tap * 2 + 1) * 512);
#pragma unroll
            for (int rr = 0; rr < 2; ++rr) {
                const int off = ((rr + ky) * HALO + kx) * PIX_STRIDE_H;  // halves
                const v8h lo = *(const v8h*)(aBase + off);       // K = c0..c0+7
                const v8h hi = *(const v8h*)(aBase + off + 16);  // K = 16+c0..
                const v16h a = __builtin_shufflevector(
                    lo, hi, 0, 1, 2, 3, 4, 5, 6, 7, 8, 9, 10, 11, 12, 13, 14, 15);
                if (rr == 0) {
                    acc00 = __builtin_amdgcn_wmma_f32_16x16x32_f16(
                        false, a, false, b0, (short)0, acc00, false, false);
                    acc01 = __builtin_amdgcn_wmma_f32_16x16x32_f16(
                        false, a, false, b1, (short)0, acc01, false, false);
                } else {
                    acc10 = __builtin_amdgcn_wmma_f32_16x16x32_f16(
                        false, a, false, b0, (short)0, acc10, false, false);
                    acc11 = __builtin_amdgcn_wmma_f32_16x16x32_f16(
                        false, a, false, b1, (short)0, acc11, false, false);
                }
            }
        }
    }

    // ---------------- Stage 3: bias + store ----------------
    const float bz0 = bsum[m];
    const float bz1 = bsum[16 + m];

    // One base pointer; all 32 stores use constant immediate offsets.
    float* ob = out + (((size_t)img * IMG_H + h0 + wv * 2) * IMG_W + w0 + 8 * laneHi)
                          * COUT + m;
#pragma unroll
    for (int rr = 0; rr < 2; ++rr) {
#pragma unroll
        for (int r = 0; r < 8; ++r) {
            const size_t o = (size_t)rr * IMG_W * COUT + (size_t)r * COUT;
            if (rr == 0) {
                ob[o]      = acc00[r] + bz0;
                ob[o + 16] = acc01[r] + bz1;
            } else {
                ob[o]      = acc10[r] + bz0;
                ob[o + 16] = acc11[r] + bz1;
            }
        }
    }
}

// ---------------------------------------------------------------------------
extern "C" void kernel_launch(void* const* d_in, const int* in_sizes, int n_in,
                              void* d_out, int out_size, void* d_ws, size_t ws_size,
                              hipStream_t stream) {
    const float* x    = (const float*)d_in[0];   // [16,256,256,32]
    const float* W    = (const float*)d_in[1];   // [4,3,3,32,32]
    const float* bias = (const float*)d_in[2];   // [4,32]
    float* out        = (float*)d_out;           // [16,256,256,32]

    _Float16* wsB = (_Float16*)d_ws;                                  // 9216 halves
    float*    bsm = (float*)((char*)d_ws + NTAPS * 2 * 32 * 16 * 2);  // 32 floats

    multibin_prep<<<36, 256, 0, stream>>>(W, bias, wsB, bsm);

    // 16 images x 16 x 16 tiles of 16x16 pixels = 4096 blocks.
    multibin_conv<<<4096, 256, 0, stream>>>(x, wsB, bsm, out);
}
